// TBlock_33131377722102
// MI455X (gfx1250) — matrix operand order repaired and checked by reading, not compiled, not run on hardware
//
#include <hip/hip_runtime.h>
#include <hip/hip_bf16.h>

// Problem sizes (fixed by the reference)
#define BB   2
#define SS   2048
#define DDIM 1024
#define HH   16
#define DHH  64
#define FFF  4096
#define MM   (BB * SS)   // 4096 rows

typedef __attribute__((ext_vector_type(8)))  __bf16 v8bf;
typedef __attribute__((ext_vector_type(16))) __bf16 v16bf;
typedef __attribute__((ext_vector_type(8)))  float  v8f;

#define WMMA_BF16(a, b, c) \
  __builtin_amdgcn_wmma_f32_16x16x32_bf16(false, (a), false, (b), (short)0, (c), false, false)

__device__ __forceinline__ v16bf cat16(v8bf lo, v8bf hi) {
  return __builtin_shufflevector(lo, hi, 0, 1, 2, 3, 4, 5, 6, 7, 8, 9, 10, 11, 12, 13, 14, 15);
}

// A fragment (16x32, M x K), contiguous K order (matches B fragments):
//   lanes 0-15 : M=lane,    elems 0..15 = K 0..15
//   lanes 16-31: M=lane-16, elems 0..15 = K 16..31
__device__ __forceinline__ v16bf load_a_frag(const __bf16* A, int lda, int row0, int k0, int lane) {
  int row = row0 + (lane & 15);
  int kb  = k0 + ((lane & 16) ? 16 : 0);
  const __bf16* p = A + (size_t)row * lda + kb;
  v8bf lo = *(const v8bf*)(p);
  v8bf hi = *(const v8bf*)(p + 8);
  return cat16(lo, hi);
}

// B fragment (32x16, K x N) from W stored [N,K] row-major (K contiguous):
//   lanes 0-15 : N=lane,    K = k0..k0+15
//   lanes 16-31: N=lane-16, K = k0+16..k0+31
__device__ __forceinline__ v16bf load_b_frag(const __bf16* W, int ldb, int n0, int k0, int lane) {
  int n  = n0 + (lane & 15);
  int kb = k0 + ((lane & 16) ? 16 : 0);
  const __bf16* p = W + (size_t)n * ldb + kb;
  v8bf lo = *(const v8bf*)(p);
  v8bf hi = *(const v8bf*)(p + 8);
  return cat16(lo, hi);
}

// Async global -> LDS copy of 16 bytes (gfx1250, tracked by ASYNCcnt).
// LDS byte offset = low 32 bits of the generic shared address (flat aperture rule).
__device__ __forceinline__ void async_copy_b128(const __bf16* gsrc, __bf16* ldst) {
  unsigned lds_off = (unsigned)(uintptr_t)ldst;
  unsigned long long ga = (unsigned long long)(uintptr_t)gsrc;
  asm volatile("global_load_async_to_lds_b128 %0, %1, off"
               :: "v"(lds_off), "v"(ga) : "memory");
}
// 128-byte contiguous async copy: INST_OFFSET is added to BOTH the LDS and the
// global address (ISA 08 async pseudocode), so one base pair serves all 8 b128s.
__device__ __forceinline__ void async_copy_row128(const __bf16* gsrc, __bf16* ldst) {
  unsigned lds_off = (unsigned)(uintptr_t)ldst;
  unsigned long long ga = (unsigned long long)(uintptr_t)gsrc;
  asm volatile(
      "global_load_async_to_lds_b128 %0, %1, off\n\t"
      "global_load_async_to_lds_b128 %0, %1, off offset:16\n\t"
      "global_load_async_to_lds_b128 %0, %1, off offset:32\n\t"
      "global_load_async_to_lds_b128 %0, %1, off offset:48\n\t"
      "global_load_async_to_lds_b128 %0, %1, off offset:64\n\t"
      "global_load_async_to_lds_b128 %0, %1, off offset:80\n\t"
      "global_load_async_to_lds_b128 %0, %1, off offset:96\n\t"
      "global_load_async_to_lds_b128 %0, %1, off offset:112"
      :: "v"(lds_off), "v"(ga) : "memory");
}
__device__ __forceinline__ void wait_async() {
  asm volatile("s_wait_asynccnt 0x0" ::: "memory");
}

// ---------------------------------------------------------------------------
// f32 -> bf16 conversion (weights)
__global__ void cvt_bf16_kernel(const float* __restrict__ in, __bf16* __restrict__ out, int n) {
  int i = blockIdx.x * 256 + threadIdx.x;
  if (i < n) out[i] = (__bf16)in[i];
}

// ---------------------------------------------------------------------------
// LayerNorm over last dim (D=1024); optional residual add; emits f32 + bf16.
__global__ void ln_kernel(const float* __restrict__ X, const float* __restrict__ R,
                          const float* __restrict__ w, const float* __restrict__ bias,
                          float* __restrict__ Yf, __bf16* __restrict__ Ybf) {
  __shared__ float buf[DDIM];
  __shared__ float red[256];
  int row = blockIdx.x;
  int t = threadIdx.x;
  const float* x = X + (size_t)row * DDIM;
  float s = 0.f;
  for (int i = t; i < DDIM; i += 256) {
    float v = x[i];
    if (R) v += R[(size_t)row * DDIM + i];
    buf[i] = v;
    s += v;
  }
  red[t] = s;
  __syncthreads();
  for (int off = 128; off > 0; off >>= 1) {
    if (t < off) red[t] += red[t + off];
    __syncthreads();
  }
  float mu = red[0] * (1.0f / DDIM);
  __syncthreads();
  float vs = 0.f;
  for (int i = t; i < DDIM; i += 256) {
    float d = buf[i] - mu;
    vs += d * d;
  }
  red[t] = vs;
  __syncthreads();
  for (int off = 128; off > 0; off >>= 1) {
    if (t < off) red[t] += red[t + off];
    __syncthreads();
  }
  float rstd = rsqrtf(red[0] * (1.0f / DDIM) + 1e-5f);
  for (int i = t; i < DDIM; i += 256) {
    float y = (buf[i] - mu) * rstd * w[i] + bias[i];
    Yf[(size_t)row * DDIM + i] = y;
    Ybf[(size_t)row * DDIM + i] = (__bf16)y;
  }
}

// ---------------------------------------------------------------------------
// C = A(bf16,[M,K]) @ W(bf16,[N,K])^T, double-buffered async-LDS staging.
// Block: 256 threads / 8 waves; block tile 64(M) x 256(N); wave tile 32x64.
// MODE 0: store bf16. MODE 1: silu -> bf16. MODE 2: f32 store of C + Res.
#define APAD 40  // padded K-stride (elems) for LDS tiles: 80B rows, 16B aligned
template <int MODE>
__global__ void gemm_bf16_kernel(const __bf16* __restrict__ A, const __bf16* __restrict__ W,
                                 void* __restrict__ Cout, const float* __restrict__ Res,
                                 int M, int N, int K) {
  __shared__ __bf16 At[2][64][APAD];    // 64 rows x 32 k, double buffered
  __shared__ __bf16 Bt[2][256][APAD];   // 256 cols x 32 k, double buffered
  int t = threadIdx.x;
  int lane = t & 31;
  int wave = t >> 5;
  int wm = wave >> 2;  // 0..1 -> 32 rows each
  int wn = wave & 3;   // 0..3 -> 64 cols each
  int bm0 = blockIdx.x * 64;
  int bn0 = blockIdx.y * 256;

  // staging assignment: thread t copies A chunk (row t/4, 8-elem chunk t%4)
  // and 4 B chunks (rows t/4 + {0,64,128,192}).
  int sr = t >> 2;            // 0..63
  int sc = (t & 3) * 8;       // 0,8,16,24
  const __bf16* Ab  = A + (size_t)(bm0 + sr) * K + sc;
  const __bf16* Wb0 = W + (size_t)(bn0 + sr) * K + sc;
  const __bf16* Wb1 = W + (size_t)(bn0 + 64 + sr) * K + sc;
  const __bf16* Wb2 = W + (size_t)(bn0 + 128 + sr) * K + sc;
  const __bf16* Wb3 = W + (size_t)(bn0 + 192 + sr) * K + sc;

  auto stage = [&](int k, int buf) {
    async_copy_b128(Ab + k,  &At[buf][sr][sc]);
    async_copy_b128(Wb0 + k, &Bt[buf][sr][sc]);
    async_copy_b128(Wb1 + k, &Bt[buf][64 + sr][sc]);
    async_copy_b128(Wb2 + k, &Bt[buf][128 + sr][sc]);
    async_copy_b128(Wb3 + k, &Bt[buf][192 + sr][sc]);
  };

  v8f acc[2][4] = {};
  int nk = K >> 5;
  stage(0, 0);
  for (int i = 0; i < nk; ++i) {
    wait_async();      // own group-i copies have landed
    __syncthreads();   // everyone's group-i copies have landed; buf[(i+1)&1] reads done
    if (i + 1 < nk) stage((i + 1) << 5, (i + 1) & 1);  // overlap with compute below
    int buf = i & 1;

    v16bf a0 = load_a_frag(&At[buf][wm * 32][0], APAD, 0, 0, lane);
    v16bf a1 = load_a_frag(&At[buf][wm * 32][0], APAD, 16, 0, lane);
    v16bf b0 = load_b_frag(&Bt[buf][wn * 64][0], APAD, 0, 0, lane);
    v16bf b1 = load_b_frag(&Bt[buf][wn * 64][0], APAD, 16, 0, lane);
    v16bf b2 = load_b_frag(&Bt[buf][wn * 64][0], APAD, 32, 0, lane);
    v16bf b3 = load_b_frag(&Bt[buf][wn * 64][0], APAD, 48, 0, lane);
    acc[0][0] = WMMA_BF16(a0, b0, acc[0][0]);
    acc[0][1] = WMMA_BF16(a0, b1, acc[0][1]);
    acc[0][2] = WMMA_BF16(a0, b2, acc[0][2]);
    acc[0][3] = WMMA_BF16(a0, b3, acc[0][3]);
    acc[1][0] = WMMA_BF16(a1, b0, acc[1][0]);
    acc[1][1] = WMMA_BF16(a1, b1, acc[1][1]);
    acc[1][2] = WMMA_BF16(a1, b2, acc[1][2]);
    acc[1][3] = WMMA_BF16(a1, b3, acc[1][3]);
  }

  int halfadd = (lane & 16) ? 8 : 0;
  int nlane = lane & 15;
  auto store_tile = [&](v8f cv, int row0, int col0) {
    for (int r = 0; r < 8; ++r) {
      int row = row0 + r + halfadd;
      size_t idx = (size_t)row * N + col0 + nlane;
      float v = cv[r];
      if (MODE == 0) {
        ((__bf16*)Cout)[idx] = (__bf16)v;
      } else if (MODE == 1) {
        float sv = v / (1.0f + __expf(-v));  // silu
        ((__bf16*)Cout)[idx] = (__bf16)sv;
      } else {
        ((float*)Cout)[idx] = v + Res[idx];
      }
    }
  };
  for (int i = 0; i < 2; ++i)
    for (int j = 0; j < 4; ++j)
      store_tile(acc[i][j], bm0 + wm * 32 + i * 16, bn0 + wn * 64 + j * 16);
}

// ---------------------------------------------------------------------------
// Flash attention, one wave = 16 queries of one (b,h). Q/K/V bf16 [B,S,H*DH].
#define VPAD 72  // padded d-stride for V slab (144B rows: 16B-aligned, bank-friendly)
__global__ void attn_kernel(const __bf16* __restrict__ Q, const __bf16* __restrict__ Km,
                            const __bf16* __restrict__ Vm, float* __restrict__ O) {
  __shared__ __bf16 pl[8][16][32];    // per-wave P staging (C-layout -> A-layout)
  __shared__ __bf16 vt[8][32][VPAD];  // per-wave V slab: 32 keys x 64 d
  int lane = threadIdx.x & 31;
  int wave = threadIdx.x >> 5;
  int task = blockIdx.x * 8 + wave;    // 0..4095
  int qt = task & 127;                 // S/16 = 128 query tiles
  int bh = task >> 7;                  // 0..31
  int h = bh & (HH - 1);
  int b = bh >> 4;
  int m0 = qt * 16;

  float slope = exp2f(-0.5f * (float)(h + 1));  // 2^-((h+1)*8/H), H=16
  const __bf16* Qh = Q + (size_t)b * SS * DDIM + h * DHH;
  const __bf16* Kh = Km + (size_t)b * SS * DDIM + h * DHH;
  const __bf16* Vh = Vm + (size_t)b * SS * DDIM + h * DHH;

  // Q fragments for d in [0,32) and [32,64); rows are sequence positions (lda = D)
  v16bf qa0 = load_a_frag(Qh, DDIM, m0, 0, lane);
  v16bf qa1 = load_a_frag(Qh, DDIM, m0, 32, lane);

  float mrow[8], lrow[8];
  v8f acc[4] = {};  // O tile 16x64 as 4 N-chunks of 16
  for (int r = 0; r < 8; ++r) { mrow[r] = -1e30f; lrow[r] = 0.f; }

  int halfadd = (lane & 16) ? 8 : 0;
  int nlane = lane & 15;
  int nblk = (m0 + 16 + 31) >> 5;  // causal: key blocks of 32 up to m0+15

  for (int blk = 0; blk < nblk; ++blk) {
    int j0 = blk * 32;

    // async-stage V slab for this key block: lane copies key row j0+lane (128B),
    // one base address pair + immediate offsets (same stride both sides).
    async_copy_row128(Vh + (size_t)(j0 + lane) * DDIM, &vt[wave][lane][0]);

    // S tile 16x32 = two 16x16 C-frags, K-dim = DH = 64 (2 WMMA k-steps each)
    v8f s0 = {}, s1 = {};
    v16bf kb;
    kb = load_b_frag(Kh, DDIM, j0, 0, lane);       s0 = WMMA_BF16(qa0, kb, s0);
    kb = load_b_frag(Kh, DDIM, j0, 32, lane);      s0 = WMMA_BF16(qa1, kb, s0);
    kb = load_b_frag(Kh, DDIM, j0 + 16, 0, lane);  s1 = WMMA_BF16(qa0, kb, s1);
    kb = load_b_frag(Kh, DDIM, j0 + 16, 32, lane); s1 = WMMA_BF16(qa1, kb, s1);

    for (int r = 0; r < 8; ++r) {
      int qrow = m0 + r + halfadd;
      int k0i = j0 + nlane;
      int k1i = k0i + 16;
      float v0 = s0[r] * 0.125f + slope * (float)(k0i - qrow);
      float v1 = s1[r] * 0.125f + slope * (float)(k1i - qrow);
      if (k0i > qrow) v0 = -1e30f;
      if (k1i > qrow) v1 = -1e30f;
      float rmax = fmaxf(v0, v1);
      for (int off = 1; off < 16; off <<= 1) rmax = fmaxf(rmax, __shfl_xor(rmax, off, 32));
      float mnew = fmaxf(mrow[r], rmax);
      float p0 = __expf(v0 - mnew);
      float p1 = __expf(v1 - mnew);
      float rsum = p0 + p1;
      for (int off = 1; off < 16; off <<= 1) rsum += __shfl_xor(rsum, off, 32);
      float alpha = __expf(mrow[r] - mnew);
      lrow[r] = lrow[r] * alpha + rsum;
      mrow[r] = mnew;
      for (int c = 0; c < 4; ++c) acc[c][r] *= alpha;
      pl[wave][r + halfadd][nlane]      = (__bf16)p0;
      pl[wave][r + halfadd][nlane + 16] = (__bf16)p1;
    }
    // reload P in A-fragment layout (same-wave DS ops are in-order)
    v16bf pfrag = load_a_frag(&pl[wave][0][0], 32, 0, 0, lane);

    // wait for this wave's async V copies, then gather B-frags from LDS
    wait_async();
    int keybase = (lane & 16) ? 16 : 0;
    for (int c = 0; c < 4; ++c) {
      int dcol = c * 16 + nlane;
      v16bf vf;
      for (int e = 0; e < 16; ++e) vf[e] = vt[wave][keybase + e][dcol];
      acc[c] = WMMA_BF16(pfrag, vf, acc[c]);
    }
  }

  for (int r = 0; r < 8; ++r) {
    float inv = 1.0f / lrow[r];
    int row = m0 + r + halfadd;
    float* op = O + ((size_t)b * SS + row) * DDIM + h * DHH + nlane;
    for (int c = 0; c < 4; ++c) op[c * 16] = acc[c][r] * inv;
  }
}

// ---------------------------------------------------------------------------
extern "C" void kernel_launch(void* const* d_in, const int* in_sizes, int n_in,
                              void* d_out, int out_size, void* d_ws, size_t ws_size,
                              hipStream_t stream) {
  (void)in_sizes; (void)n_in; (void)out_size; (void)ws_size;
  const float* x    = (const float*)d_in[0];
  const float* wq   = (const float*)d_in[1];
  const float* wk   = (const float*)d_in[2];
  const float* wv   = (const float*)d_in[3];
  const float* ln1w = (const float*)d_in[4];
  const float* ln1b = (const float*)d_in[5];
  const float* ln2w = (const float*)d_in[6];
  const float* ln2b = (const float*)d_in[7];
  const float* w1   = (const float*)d_in[8];
  const float* w2   = (const float*)d_in[9];

  char* ws = (char*)d_ws;
  size_t off = 0;
  auto alloc = [&](size_t bytes) -> void* {
    void* p = ws + off;
    off += (bytes + 255) & ~(size_t)255;
    return p;
  };
  float*  h_f   = (float*)alloc((size_t)MM * DDIM * 4);
  __bf16* h_bf  = (__bf16*)alloc((size_t)MM * DDIM * 2);
  __bf16* wq_bf = (__bf16*)alloc((size_t)DDIM * DDIM * 2);
  __bf16* wk_bf = (__bf16*)alloc((size_t)DDIM * DDIM * 2);
  __bf16* wv_bf = (__bf16*)alloc((size_t)DDIM * DDIM * 2);
  __bf16* w1_bf = (__bf16*)alloc((size_t)FFF * DDIM * 2);
  __bf16* w2_bf = (__bf16*)alloc((size_t)DDIM * FFF * 2);
  __bf16* q_bf  = (__bf16*)alloc((size_t)MM * DDIM * 2);
  __bf16* k_bf  = (__bf16*)alloc((size_t)MM * DDIM * 2);
  __bf16* v_bf  = (__bf16*)alloc((size_t)MM * DDIM * 2);
  float*  o_f   = (float*)alloc((size_t)MM * DDIM * 4);
  float*  h2_f  = (float*)alloc((size_t)MM * DDIM * 4);
  __bf16* h2_bf = (__bf16*)alloc((size_t)MM * DDIM * 2);
  __bf16* y1_bf = (__bf16*)alloc((size_t)MM * FFF * 2);

  // weight conversions
  cvt_bf16_kernel<<<(DDIM * DDIM) / 256, 256, 0, stream>>>(wq, wq_bf, DDIM * DDIM);
  cvt_bf16_kernel<<<(DDIM * DDIM) / 256, 256, 0, stream>>>(wk, wk_bf, DDIM * DDIM);
  cvt_bf16_kernel<<<(DDIM * DDIM) / 256, 256, 0, stream>>>(wv, wv_bf, DDIM * DDIM);
  cvt_bf16_kernel<<<(FFF * DDIM) / 256, 256, 0, stream>>>(w1, w1_bf, FFF * DDIM);
  cvt_bf16_kernel<<<(DDIM * FFF) / 256, 256, 0, stream>>>(w2, w2_bf, DDIM * FFF);

  // LN1: h = LN(x)
  ln_kernel<<<MM, 256, 0, stream>>>(x, nullptr, ln1w, ln1b, h_f, h_bf);

  // QKV projections: N = 1024 -> grid.y = 4
  dim3 gqkv(MM / 64, DDIM / 256);
  gemm_bf16_kernel<0><<<gqkv, 256, 0, stream>>>(h_bf, wq_bf, (void*)q_bf, nullptr, MM, DDIM, DDIM);
  gemm_bf16_kernel<0><<<gqkv, 256, 0, stream>>>(h_bf, wk_bf, (void*)k_bf, nullptr, MM, DDIM, DDIM);
  gemm_bf16_kernel<0><<<gqkv, 256, 0, stream>>>(h_bf, wv_bf, (void*)v_bf, nullptr, MM, DDIM, DDIM);

  // attention: 4096 wave-tasks, 8 waves per block
  attn_kernel<<<(BB * HH * (SS / 16)) / 8, 256, 0, stream>>>(q_bf, k_bf, v_bf, o_f);

  // LN2: h2 = LN(h + o)
  ln_kernel<<<MM, 256, 0, stream>>>(h_f, o_f, ln2w, ln2b, h2_f, h2_bf);

  // MLP: y1 = silu(h2 @ w1^T); out = h2 + y1 @ w2^T
  gemm_bf16_kernel<1><<<dim3(MM / 64, FFF / 256), 256, 0, stream>>>(h2_bf, w1_bf, (void*)y1_bf,
                                                                    nullptr, MM, FFF, DDIM);
  gemm_bf16_kernel<2><<<dim3(MM / 64, DDIM / 256), 256, 0, stream>>>(y1_bf, w2_bf, d_out, h2_f,
                                                                     MM, DDIM, FFF);
}
